// GroupedQueryAttention_25580825215367
// MI455X (gfx1250) — matrix operand order repaired and checked by reading, not compiled, not run on hardware
//
#include <hip/hip_runtime.h>
#include <cstdint>

// ---------------- problem constants ----------------
constexpr int Bsz = 2, Ssz = 2048, Dsz = 2048, Hn = 16, KVn = 4, HDs = 128;
constexpr int Rg = Hn / KVn;                 // GQA group size = 4
constexpr float THETA = 500000.0f;

#define USE_ASYNC_LDS 1

// ---------------- WMMA types ----------------
typedef __attribute__((ext_vector_type(16))) __bf16 v16bf;
typedef __attribute__((ext_vector_type(8)))  float  v8f;
typedef __attribute__((ext_vector_type(4)))  unsigned int v4u;

union Frag16 { v16bf v; v4u q[2]; };

__device__ __forceinline__ v8f wmma_bf16(v16bf a, v16bf b, v8f c) {
  // D = A(16x32 bf16) * B(32x16 bf16) + C(16x16 f32)
  return __builtin_amdgcn_wmma_f32_16x16x32_bf16(false, a, false, b, (short)0, c,
                                                 false, false);
}

__device__ __forceinline__ unsigned short f2bf(float f) {
  union { float f; unsigned u; } v; v.f = f;
  unsigned r = v.u + 0x7FFFu + ((v.u >> 16) & 1u);   // round-to-nearest-even
  return (unsigned short)(r >> 16);
}

__device__ __forceinline__ void wait_ds0() {
#if __has_builtin(__builtin_amdgcn_s_wait_dscnt)
  __builtin_amdgcn_s_wait_dscnt(0);
#else
  asm volatile("s_wait_dscnt 0" ::: "memory");
#endif
}

__device__ __forceinline__ void wait_async0() {
#if __has_builtin(__builtin_amdgcn_s_wait_asynccnt)
  __builtin_amdgcn_s_wait_asynccnt(0);
#else
  asm volatile("s_wait_asynccnt 0" ::: "memory");
#endif
}

// Async DMA of 16 bytes global -> LDS (ISA §15.18, tracked by ASYNCcnt).
// lds_byte_addr = low 32 bits of the flat shared-memory address (LDS aperture
// maps addr[31:0] to the LDS offset).
__device__ __forceinline__ void async_g2l_b128(unsigned lds_byte_addr,
                                               const void* gaddr) {
  asm volatile("global_load_async_to_lds_b128 %0, %1, off"
               :: "v"(lds_byte_addr), "v"(gaddr)
               : "memory");
}

// ---------------- elementwise converters ----------------
__global__ void cvt_f32_to_bf16(const float* __restrict__ in,
                                unsigned short* __restrict__ out, int n) {
  for (int i = blockIdx.x * blockDim.x + threadIdx.x; i < n;
       i += gridDim.x * blockDim.x)
    out[i] = f2bf(in[i]);
}

// out[n][k] = in[k][n]  (weight [K,N] f32 -> [N,K] bf16)
__global__ void transpose_to_bf16(const float* __restrict__ in,
                                  unsigned short* __restrict__ out,
                                  int Kd, int N) {
  int total = Kd * N;
  for (int i = blockIdx.x * blockDim.x + threadIdx.x; i < total;
       i += gridDim.x * blockDim.x) {
    int k = i / N, c = i - k * N;
    out[(size_t)c * Kd + k] = f2bf(in[i]);
  }
}

// ---------------- block-cooperative double-buffered WMMA GEMM ----------------
// C[M,N] = A[M,K] * Bt[N,K]^T.  Block = 256 thr (8 waves) -> 128x128 C tile.
// Wave (wmid 0..3, wnid 0..1) -> 32x64 sub-tile = 2x4 WMMA tiles.
// LDS rows padded to 40 ushorts (80B) => 20-bank stride, conflict-free.
constexpr int LPAD = 40;

__device__ __forceinline__ void stage_tile(const unsigned short* __restrict__ G,
                                           int Kd, int k0,
                                           unsigned short* lbuf, int tid) {
  const int r = tid >> 1, half = tid & 1;          // 128 rows x two 32B halves
  const unsigned short* g = G + (size_t)r * Kd + k0 + half * 16;
#if USE_ASYNC_LDS
  const unsigned la = (unsigned)(uintptr_t)lbuf + (unsigned)(r * (LPAD * 2) + half * 32);
  async_g2l_b128(la, g);
  async_g2l_b128(la + 16, g + 8);
#else
  unsigned short* l = lbuf + r * LPAD + half * 16;
  *(v4u*)(l)     = *(const v4u*)(g);
  *(v4u*)(l + 8) = *(const v4u*)(g + 8);
#endif
}

__device__ __forceinline__ void compute_step(const unsigned short* lA,
                                             const unsigned short* lB,
                                             int wmid, int wnid, int hh, int ln,
                                             v8f acc[2][4]) {
  Frag16 af[2], bfr[4];
#pragma unroll
  for (int mt = 0; mt < 2; ++mt) {
    const unsigned short* p = lA + (wmid * 32 + mt * 16 + ln) * LPAD;
    af[mt].q[0] = *(const v4u*)(p + hh * 16);
    af[mt].q[1] = *(const v4u*)(p + hh * 16 + 8);
  }
#pragma unroll
  for (int nt = 0; nt < 4; ++nt) {
    const unsigned short* p = lB + (wnid * 64 + nt * 16 + ln) * LPAD + hh * 16;
    bfr[nt].q[0] = *(const v4u*)(p);
    bfr[nt].q[1] = *(const v4u*)(p + 8);
  }
#pragma unroll
  for (int mt = 0; mt < 2; ++mt)
#pragma unroll
    for (int nt = 0; nt < 4; ++nt)
      acc[mt][nt] = wmma_bf16(af[mt].v, bfr[nt].v, acc[mt][nt]);
}

__global__ void __launch_bounds__(256)
gemm_bf16_nt(const unsigned short* __restrict__ A,
             const unsigned short* __restrict__ Bt,
             float* __restrict__ C, int M, int N, int Kd) {
  __shared__ __align__(16) unsigned short lA[2][128 * LPAD];
  __shared__ __align__(16) unsigned short lB[2][128 * LPAD];

  const int tid  = threadIdx.x;
  const int lane = tid & 31, wid = tid >> 5;
  const int wmid = wid & 3, wnid = wid >> 2;       // 4 waves in M, 2 in N
  const int hh = lane >> 4, ln = lane & 15;
  const int n0 = blockIdx.x * 128, m0 = blockIdx.y * 128;

  const unsigned short* Ab = A  + (size_t)m0 * Kd;
  const unsigned short* Bb = Bt + (size_t)n0 * Kd;

  const v8f vz = {0.f,0.f,0.f,0.f,0.f,0.f,0.f,0.f};
  v8f acc[2][4] = {{vz,vz,vz,vz},{vz,vz,vz,vz}};

  stage_tile(Ab, Kd, 0, lA[0], tid);
  stage_tile(Bb, Kd, 0, lB[0], tid);
  wait_async0();
  __syncthreads();

  int buf = 0;
  for (int k0 = 32; k0 < Kd; k0 += 32) {
    // prefetch next K-slab into the other buffer (async, overlapped with WMMA)
    stage_tile(Ab, Kd, k0, lA[buf ^ 1], tid);
    stage_tile(Bb, Kd, k0, lB[buf ^ 1], tid);
    compute_step(lA[buf], lB[buf], wmid, wnid, hh, ln, acc);
    wait_async0();
    __syncthreads();
    buf ^= 1;
  }
  compute_step(lA[buf], lB[buf], wmid, wnid, hh, ln, acc);

#pragma unroll
  for (int mt = 0; mt < 2; ++mt)
#pragma unroll
    for (int nt = 0; nt < 4; ++nt)
#pragma unroll
      for (int j = 0; j < 8; ++j) {
        int row = m0 + wmid * 32 + mt * 16 + hh * 8 + j;
        int col = n0 + wnid * 64 + nt * 16 + ln;
        C[(size_t)row * N + col] = acc[mt][nt][j];
      }
}

// ---------------- RoPE + layout packing ----------------
__global__ void rope_q_kernel(const float* __restrict__ qf,
                              const int* __restrict__ pos,
                              unsigned short* __restrict__ Qbf) {
  const int n = Bsz * Ssz * Hn * 64;              // 2^22
  for (int i = blockIdx.x * blockDim.x + threadIdx.x; i < n;
       i += gridDim.x * blockDim.x) {
    int hd = i & 63;
    int h  = (i >> 6) & (Hn - 1);
    int s  = (i >> 10) & (Ssz - 1);
    int b  = i >> 21;
    float p   = (float)pos[b * Ssz + s];
    float inv = __powf(THETA, -(float)hd * (1.0f / 64.0f));
    float sn, c; __sincosf(p * inv, &sn, &c);
    const float* src = qf + (size_t)(b * Ssz + s) * (Hn * HDs) + h * HDs + hd;
    float q1 = src[0], q2 = src[64];
    unsigned short* dst = Qbf + ((size_t)(b * Hn + h) * Ssz + s) * HDs + hd;
    dst[0]  = f2bf(q1 * c - q2 * sn);
    dst[64] = f2bf(q2 * c + q1 * sn);
  }
}

__global__ void rope_kv_kernel(const float* __restrict__ kf,
                               const float* __restrict__ vf,
                               const int* __restrict__ pos,
                               unsigned short* __restrict__ Kbf,
                               unsigned short* __restrict__ Vt) {
  const int n = Bsz * Ssz * KVn * 64;             // 2^20
  for (int i = blockIdx.x * blockDim.x + threadIdx.x; i < n;
       i += gridDim.x * blockDim.x) {
    int hd = i & 63;
    int kv = (i >> 6) & (KVn - 1);
    int s  = (i >> 8) & (Ssz - 1);
    int b  = i >> 19;
    float p   = (float)pos[b * Ssz + s];
    float inv = __powf(THETA, -(float)hd * (1.0f / 64.0f));
    float sn, c; __sincosf(p * inv, &sn, &c);
    size_t srcb = (size_t)(b * Ssz + s) * (KVn * HDs) + kv * HDs + hd;
    float k1 = kf[srcb], k2 = kf[srcb + 64];
    unsigned short* kd = Kbf + ((size_t)(b * KVn + kv) * Ssz + s) * HDs + hd;
    kd[0]  = f2bf(k1 * c - k2 * sn);
    kd[64] = f2bf(k2 * c + k1 * sn);
    float v1 = vf[srcb], v2 = vf[srcb + 64];
    unsigned short* vb = Vt + ((size_t)(b * KVn + kv) * HDs) * Ssz;
    vb[(size_t)hd * Ssz + s]        = f2bf(v1);
    vb[(size_t)(hd + 64) * Ssz + s] = f2bf(v2);
  }
}

// ---------------- flash attention (one wave = 16 q rows of one head) ----------------
__global__ void attn_kernel(const unsigned short* __restrict__ Q,
                            const unsigned short* __restrict__ Kb,
                            const unsigned short* __restrict__ Vt,
                            unsigned short* __restrict__ Obf) {
  __shared__ __align__(16) unsigned short plds[4][16 * 32]; // per-wave P tile
  const int wid  = threadIdx.x >> 5;
  const int lane = threadIdx.x & 31;
  const int gw   = blockIdx.x * 4 + wid;
  const int QBLKS = Ssz / 16;
  const int qblk = gw % QBLKS;
  const int h    = (gw / QBLKS) % Hn;
  const int b    = gw / (QBLKS * Hn);
  const int kv   = h / Rg;
  const int s0   = qblk * 16;
  const int hh = lane >> 4, ln = lane & 15;
  const float scale = 0.08838834764831845f;      // 1/sqrt(128)

  Frag16 qf[4];
  const unsigned short* qrow = Q + ((size_t)(b * Hn + h) * Ssz + s0 + ln) * HDs;
#pragma unroll
  for (int kk = 0; kk < 4; ++kk) {
    qf[kk].q[0] = *(const v4u*)(qrow + kk * 32 + hh * 16);
    qf[kk].q[1] = *(const v4u*)(qrow + kk * 32 + hh * 16 + 8);
  }

  const v8f vz = {0.f,0.f,0.f,0.f,0.f,0.f,0.f,0.f};
  v8f acc[8] = {vz, vz, vz, vz, vz, vz, vz, vz};
  float mj[8], lj[8];
#pragma unroll
  for (int j = 0; j < 8; ++j) { mj[j] = -3.0e38f; lj[j] = 0.f; }

  const unsigned short* Kbase = Kb + (size_t)(b * KVn + kv) * Ssz * HDs;
  const unsigned short* Vbase = Vt + (size_t)(b * KVn + kv) * HDs * Ssz;
  unsigned short* pl = plds[wid];

  for (int t0 = 0; t0 < s0 + 16; t0 += 32) {
    // prefetch next K tile row for this lane (global_prefetch_b8)
    __builtin_prefetch((const void*)(Kbase + (size_t)(t0 + 32 + ln) * HDs), 0, 0);

    v8f c0 = vz, c1 = vz;
#pragma unroll
    for (int sub = 0; sub < 2; ++sub) {
      const unsigned short* krow = Kbase + (size_t)(t0 + sub * 16 + ln) * HDs;
#pragma unroll
      for (int kk = 0; kk < 4; ++kk) {
        Frag16 kfr;
        kfr.q[0] = *(const v4u*)(krow + kk * 32 + hh * 16);
        kfr.q[1] = *(const v4u*)(krow + kk * 32 + hh * 16 + 8);
        if (sub == 0) c0 = wmma_bf16(qf[kk].v, kfr.v, c0);
        else          c1 = wmma_bf16(qf[kk].v, kfr.v, c1);
      }
    }
#pragma unroll
    for (int j = 0; j < 8; ++j) {
      const int srow = s0 + hh * 8 + j;
      c0[j] = (t0 + ln      <= srow) ? c0[j] * scale : -3.0e38f;
      c1[j] = (t0 + 16 + ln <= srow) ? c1[j] * scale : -3.0e38f;
      float mx = fmaxf(c0[j], c1[j]);
#pragma unroll
      for (int o = 8; o >= 1; o >>= 1) mx = fmaxf(mx, __shfl_xor(mx, o, 32));
      const float mnew = fmaxf(mj[j], mx);
      const float corr = __expf(mj[j] - mnew);
      mj[j] = mnew;
      c0[j] = __expf(c0[j] - mnew);
      c1[j] = __expf(c1[j] - mnew);
      float rs = c0[j] + c1[j];
#pragma unroll
      for (int o = 8; o >= 1; o >>= 1) rs += __shfl_xor(rs, o, 32);
      lj[j] = lj[j] * corr + rs;
#pragma unroll
      for (int nt = 0; nt < 8; ++nt) acc[nt][j] *= corr;
    }
#pragma unroll
    for (int j = 0; j < 8; ++j) {
      pl[(hh * 8 + j) * 32 + ln]      = f2bf(c0[j]);
      pl[(hh * 8 + j) * 32 + 16 + ln] = f2bf(c1[j]);
    }
    wait_ds0();
    Frag16 pf;
    const unsigned short* prow = pl + ln * 32;
    pf.q[0] = *(const v4u*)(prow + hh * 16);
    pf.q[1] = *(const v4u*)(prow + hh * 16 + 8);
#pragma unroll
    for (int nt = 0; nt < 8; ++nt) {
      const unsigned short* vrow = Vbase + (size_t)(nt * 16 + ln) * Ssz;
      Frag16 vfr;
      vfr.q[0] = *(const v4u*)(vrow + t0 + hh * 16);
      vfr.q[1] = *(const v4u*)(vrow + t0 + hh * 16 + 8);
      acc[nt] = wmma_bf16(pf.v, vfr.v, acc[nt]);
    }
  }
  float invl[8];
#pragma unroll
  for (int j = 0; j < 8; ++j) invl[j] = 1.0f / lj[j];
#pragma unroll
  for (int nt = 0; nt < 8; ++nt)
#pragma unroll
    for (int j = 0; j < 8; ++j) {
      const int srow = s0 + hh * 8 + j;
      const int col  = h * HDs + nt * 16 + ln;
      Obf[((size_t)b * Ssz + srow) * (Hn * HDs) + col] =
          f2bf(acc[nt][j] * invl[j]);
    }
}

// ---------------- host-side launcher ----------------
extern "C" void kernel_launch(void* const* d_in, const int* in_sizes, int n_in,
                              void* d_out, int out_size, void* d_ws, size_t ws_size,
                              hipStream_t stream) {
  const float* x  = (const float*)d_in[0];
  const float* wq = (const float*)d_in[1];
  const float* wk = (const float*)d_in[2];
  const float* wv = (const float*)d_in[3];
  const float* wo = (const float*)d_in[4];
  // d_in[5] = attn_mask: pure causal, applied analytically in-kernel
  const int* pos  = (const int*)d_in[6];
  float* out = (float*)d_out;

  const int M = Bsz * Ssz;            // 4096 token rows
  char* w = (char*)d_ws;
  auto alloc = [&](size_t bytes) -> char* {
    char* p = w; w += (bytes + 255) & ~(size_t)255; return p;
  };
  unsigned short* xbf  = (unsigned short*)alloc((size_t)M * Dsz * 2);
  unsigned short* wqT  = (unsigned short*)alloc((size_t)Dsz * (Hn * HDs) * 2);
  unsigned short* wkT  = (unsigned short*)alloc((size_t)Dsz * (KVn * HDs) * 2);
  unsigned short* wvT  = (unsigned short*)alloc((size_t)Dsz * (KVn * HDs) * 2);
  unsigned short* woT  = (unsigned short*)alloc((size_t)(Hn * HDs) * Dsz * 2);
  float*          qf32 = (float*)alloc((size_t)M * (Hn * HDs) * 4);
  float*          kf32 = (float*)alloc((size_t)M * (KVn * HDs) * 4);
  float*          vf32 = (float*)alloc((size_t)M * (KVn * HDs) * 4);
  unsigned short* Qbf  = (unsigned short*)alloc((size_t)Bsz * Hn * Ssz * HDs * 2);
  unsigned short* Kbf  = (unsigned short*)alloc((size_t)Bsz * KVn * Ssz * HDs * 2);
  unsigned short* Vtbf = (unsigned short*)alloc((size_t)Bsz * KVn * HDs * Ssz * 2);
  unsigned short* Obf  = (unsigned short*)alloc((size_t)M * (Hn * HDs) * 2);

  // 1) precision conversion + weight transposes
  cvt_f32_to_bf16<<<2048, 256, 0, stream>>>(x, xbf, M * Dsz);
  transpose_to_bf16<<<2048, 256, 0, stream>>>(wq, wqT, Dsz, Hn * HDs);
  transpose_to_bf16<<<1024, 256, 0, stream>>>(wk, wkT, Dsz, KVn * HDs);
  transpose_to_bf16<<<1024, 256, 0, stream>>>(wv, wvT, Dsz, KVn * HDs);
  transpose_to_bf16<<<2048, 256, 0, stream>>>(wo, woT, Hn * HDs, Dsz);

  // 2) QKV projections (async-LDS double-buffered WMMA GEMM)
  gemm_bf16_nt<<<dim3((Hn * HDs) / 128, M / 128), 256, 0, stream>>>(
      xbf, wqT, qf32, M, Hn * HDs, Dsz);
  gemm_bf16_nt<<<dim3((KVn * HDs) / 128, M / 128), 256, 0, stream>>>(
      xbf, wkT, kf32, M, KVn * HDs, Dsz);
  gemm_bf16_nt<<<dim3((KVn * HDs) / 128, M / 128), 256, 0, stream>>>(
      xbf, wvT, vf32, M, KVn * HDs, Dsz);

  // 3) RoPE + attention-friendly layouts (V transposed)
  rope_q_kernel<<<2048, 256, 0, stream>>>(qf32, pos, Qbf);
  rope_kv_kernel<<<1024, 256, 0, stream>>>(kf32, vf32, pos, Kbf, Vtbf);

  // 4) causal flash attention (WMMA QK^T and PV)
  attn_kernel<<<Bsz * Hn * (Ssz / 16) / 4, 128, 0, stream>>>(Qbf, Kbf, Vtbf, Obf);

  // 5) output projection -> fp32 d_out
  gemm_bf16_nt<<<dim3(Dsz / 128, M / 128), 256, 0, stream>>>(
      Obf, woT, out, M, Dsz, Hn * HDs);
}
